// CrossAttentionBlockPatched_3264175145030
// MI455X (gfx1250) — compile-verified, matrix operand order checked by reading, not run on hardware
//
#include <hip/hip_runtime.h>
#include <hip/hip_bf16.h>
#include <math.h>

// ---------------------------------------------------------------------------
// CrossAttentionBlock (cosine attention, tau=2) for MI455X / gfx1250.
// All GEMMs via v_wmma_f32_16x16x32_bf16 (f32 accumulate); GEMM tiles staged
// through LDS with double-buffered gfx1250 async global->LDS copies
// (ASYNCcnt pipelining); flash-style attention (no materialized scores);
// f32 statistics for LN/softmax/L2norm.
// ---------------------------------------------------------------------------

#define DIMD   1024
#define HEADS  16
#define HD     64
#define BATCH  4
#define LQ     512
#define LKV    2048
#define MQ     (BATCH * LQ)    // 2048 query rows
#define MKV    (BATCH * LKV)   // 8192 kv rows
#define HID    4096

typedef __attribute__((ext_vector_type(8)))  float  v8f;
typedef __attribute__((ext_vector_type(8)))  __bf16 v8bf;
typedef __attribute__((ext_vector_type(16))) __bf16 v16bf;

// Per-lane WMMA 16-bit operand fragment load (A-pattern; B uses the same
// pattern against the K-contiguous transposed operand).
// lane&15 = row, lane>>4 selects K half: {0..7,16..23} vs {8..15,24..31}.
// Caller passes p = rowbase + (lane>>4)*8 ; we read p[0..7] and p[16..23].
__device__ __forceinline__ v16bf ldfrag(const __bf16* p) {
  v8bf lo = *(const v8bf*)(p);
  v8bf hi = *(const v8bf*)(p + 16);
  return __builtin_shufflevector(lo, hi, 0, 1, 2, 3, 4, 5, 6, 7,
                                         8, 9, 10, 11, 12, 13, 14, 15);
}

__device__ __forceinline__ v8f wmma_bf16(v16bf a, v16bf b, v8f c) {
  return __builtin_amdgcn_wmma_f32_16x16x32_bf16(
      /*neg_a=*/false, a, /*neg_b=*/false, b,
      /*c_mod=*/(short)0, c, /*reuse_a=*/false, /*reuse_b=*/false);
}

// gfx1250 async global->LDS copy of 16 bytes, tracked by ASYNCcnt.
// lds_off: LDS byte offset (low 32 bits of generic pointer to __shared__).
__device__ __forceinline__ void async_copy_b128(unsigned lds_off,
                                                const void* gaddr) {
  asm volatile("global_load_async_to_lds_b128 %0, %1, off"
               :: "v"(lds_off), "v"((unsigned long long)(size_t)gaddr)
               : "memory");
}

// ---------------------------------------------------------------------------
// LayerNorm over rows of `ncols`, f32 stats, bf16 output.
// ---------------------------------------------------------------------------
__global__ void ln_rows_bf16(const float* __restrict__ x,
                             const float* __restrict__ w,
                             const float* __restrict__ bia,
                             __bf16* __restrict__ y, int ncols) {
  const int row = blockIdx.x;
  const float* xr = x + (size_t)row * ncols;
  __shared__ float red[64];
  float s = 0.f, ss = 0.f;
  for (int c = threadIdx.x; c < ncols; c += blockDim.x) {
    float v = xr[c];
    s += v;
    ss += v * v;
  }
  for (int o = 16; o > 0; o >>= 1) {
    s  += __shfl_down(s, o, 32);
    ss += __shfl_down(ss, o, 32);
  }
  const int wid = threadIdx.x >> 5;
  if ((threadIdx.x & 31) == 0) { red[wid * 2] = s; red[wid * 2 + 1] = ss; }
  __syncthreads();
  if (threadIdx.x == 0) {
    float a = 0.f, b = 0.f;
    const int nw = blockDim.x >> 5;
    for (int i = 0; i < nw; ++i) { a += red[2 * i]; b += red[2 * i + 1]; }
    red[0] = a; red[1] = b;
  }
  __syncthreads();
  const float mu  = red[0] / ncols;
  const float var = red[1] / ncols - mu * mu;
  const float inv = rsqrtf(var + 1e-6f);
  __bf16* yr = y + (size_t)row * ncols;
  for (int c = threadIdx.x; c < ncols; c += blockDim.x)
    yr[c] = (__bf16)((xr[c] - mu) * inv * w[c] + bia[c]);
}

// ---------------------------------------------------------------------------
// fp32 -> bf16 conversion (weights).
// ---------------------------------------------------------------------------
__global__ void f32_to_bf16(const float* __restrict__ x,
                            __bf16* __restrict__ y, size_t n) {
  size_t i = (size_t)blockIdx.x * blockDim.x + threadIdx.x;
  if (i < n) y[i] = (__bf16)x[i];
}

// ---------------------------------------------------------------------------
// Per-64-chunk L2 normalize (head vectors), optional post-scale (1/tau for q).
// ---------------------------------------------------------------------------
__global__ void l2norm_bf16(const float* __restrict__ x,
                            __bf16* __restrict__ y, int nchunks,
                            float postscale) {
  int i = blockIdx.x * blockDim.x + threadIdx.x;
  if (i >= nchunks) return;
  const float* p = x + (size_t)i * HD;
  float s = 0.f;
  #pragma unroll 8
  for (int j = 0; j < HD; ++j) s += p[j] * p[j];
  const float inv = postscale / fmaxf(sqrtf(s), 1e-6f);
  __bf16* q = y + (size_t)i * HD;
  #pragma unroll 8
  for (int j = 0; j < HD; ++j) q[j] = (__bf16)(p[j] * inv);
}

// ---------------------------------------------------------------------------
// V transpose: v[B*LKV, DIM] f32 -> vT[B, H, HD, LKV] bf16 so that the
// P@V B-operand is K(=kv)-contiguous.
// ---------------------------------------------------------------------------
__global__ void v_transpose_bf16(const float* __restrict__ v,
                                 __bf16* __restrict__ vT) {
  int idx = blockIdx.x * blockDim.x + threadIdx.x;
  const int total = BATCH * HEADS * HD * LKV;
  if (idx >= total) return;
  int kv = idx % LKV;
  int t = idx / LKV;
  int d = t % HD;  t /= HD;
  int h = t % HEADS;
  int b = t / HEADS;
  vT[idx] = (__bf16)v[(size_t)(b * LKV + kv) * DIMD + h * HD + d];
}

// ---------------------------------------------------------------------------
// 4-wave WMMA GEMM with double-buffered async-LDS staging:
//   C[M,N] = A[M,K](bf16) @ W[N,K](bf16)^T
// Workgroup = 128 threads = 4 waves, 64x64 output tile; each wave owns a
// 32x32 sub-tile (2x2 WMMA accumulators). Per K-step of 32, both 64x32 tiles
// are staged into LDS via global_load_async_to_lds_b128 (ASYNCcnt). The next
// K-tile's copies are issued before waiting on the current one
// (s_wait_asynccnt 4), overlapping DMA with v_wmma + ds_load_b128.
// MODE 0: Cf = C                    (f32)
// MODE 1: Cf = res + alpha*(C+bias) (f32)
// MODE 2: Cb = gelu(C+bias)         (bf16, exact erf GELU)
// ---------------------------------------------------------------------------
template <int MODE>
__global__ __launch_bounds__(128) void gemm_bf16_wmma(
    const __bf16* __restrict__ A, const __bf16* __restrict__ W,
    const float* __restrict__ bias, const float* __restrict__ res,
    const float* __restrict__ alpha, float* __restrict__ Cf,
    __bf16* __restrict__ Cb, int M, int N, int K) {
  __shared__ __bf16 smA[2][64 * 32];   // A tiles: 2 x (64 rows x 32 K), 8 KB
  __shared__ __bf16 smB[2][64 * 32];   // W tiles: 2 x (64 rows x 32 K), 8 KB

  const int tid  = threadIdx.x;
  const int lane = tid & 31;
  const int wave = tid >> 5;
  const int r = lane & 15, kh = lane >> 4;
  const int n0 = blockIdx.x * 64, m0 = blockIdx.y * 64;
  const int wm = (wave >> 1) * 32;  // wave's m-offset inside 64-tile
  const int wn = (wave & 1) * 32;   // wave's n-offset inside 64-tile

  // LDS byte offsets (generic-pointer low 32 bits == LDS offset per aperture)
  const unsigned ldsA[2] = {(unsigned)(size_t)&smA[0][0],
                            (unsigned)(size_t)&smA[1][0]};
  const unsigned ldsB[2] = {(unsigned)(size_t)&smB[0][0],
                            (unsigned)(size_t)&smB[1][0]};

  // Staging assignment: 256 16-byte chunks per tile, 2 per thread.
  const int c0row = tid >> 2,         c0seg = tid & 3;
  const int c1row = (tid + 128) >> 2, c1seg = tid & 3;

  auto stage = [&](int k, int buf) {
    async_copy_b128(ldsA[buf] + (unsigned)(c0row * 32 + c0seg * 8) * 2,
                    A + (size_t)(m0 + c0row) * K + k + c0seg * 8);
    async_copy_b128(ldsB[buf] + (unsigned)(c0row * 32 + c0seg * 8) * 2,
                    W + (size_t)(n0 + c0row) * K + k + c0seg * 8);
    async_copy_b128(ldsA[buf] + (unsigned)(c1row * 32 + c1seg * 8) * 2,
                    A + (size_t)(m0 + c1row) * K + k + c1seg * 8);
    async_copy_b128(ldsB[buf] + (unsigned)(c1row * 32 + c1seg * 8) * 2,
                    W + (size_t)(n0 + c1row) * K + k + c1seg * 8);
  };

  v8f acc00 = {}, acc01 = {}, acc10 = {}, acc11 = {};

  stage(0, 0);
  int cur = 0;
  for (int k = 0; k < K; k += 32) {
    const bool more = (k + 32) < K;
    if (more) {
      stage(k + 32, cur ^ 1);  // prefetch next tile into alternate buffer
      // Wait only until the previous group of 4 copies has landed.
      asm volatile("s_wait_asynccnt 4" ::: "memory");
    } else {
      asm volatile("s_wait_asynccnt 0" ::: "memory");
    }
    __syncthreads();  // all waves' copies for `cur` are complete + visible

    // ---- fragments from LDS (row stride 32 elems; frag spans the row) ----
    const __bf16* a0 = &smA[cur][(wm + r) * 32 + kh * 8];
    const __bf16* a1 = a0 + 16 * 32;
    const __bf16* b0 = &smB[cur][(wn + r) * 32 + kh * 8];
    const __bf16* b1 = b0 + 16 * 32;
    v16bf fa0 = ldfrag(a0);
    v16bf fa1 = ldfrag(a1);
    v16bf fb0 = ldfrag(b0);
    v16bf fb1 = ldfrag(b1);
    acc00 = wmma_bf16(fa0, fb0, acc00);
    acc01 = wmma_bf16(fa0, fb1, acc01);
    acc10 = wmma_bf16(fa1, fb0, acc10);
    acc11 = wmma_bf16(fa1, fb1, acc11);
    __syncthreads();  // done reading `cur`; it may be overwritten next iter
    cur ^= 1;
  }

  float al = 0.f;
  if (MODE == 1) al = alpha[0];

  #pragma unroll
  for (int t = 0; t < 2; ++t) {
    #pragma unroll
    for (int u = 0; u < 2; ++u) {
      v8f a = (t == 0) ? ((u == 0) ? acc00 : acc01)
                       : ((u == 0) ? acc10 : acc11);
      const int col = n0 + wn + u * 16 + r;
      #pragma unroll
      for (int e = 0; e < 8; ++e) {
        const int row = m0 + wm + t * 16 + e + 8 * kh;  // C-frag row mapping
        const size_t idx = (size_t)row * N + col;
        const float v = a[e];
        if (MODE == 0) {
          Cf[idx] = v;
        } else if (MODE == 1) {
          Cf[idx] = res[idx] + al * (v + bias[col]);
        } else {
          const float x = v + bias[col];
          Cb[idx] = (__bf16)(0.5f * x * (1.f + erff(x * 0.70710678118f)));
        }
      }
    }
  }
}

// ---------------------------------------------------------------------------
// Flash attention: one wave per (b, h, 16-row q tile). Streams kv in chunks
// of 32, online softmax in f32, P transposed through LDS to A-fragment form,
// ctx accumulated with WMMA against vT. Output ctx in bf16 [MQ, DIM].
// ---------------------------------------------------------------------------
__global__ __launch_bounds__(32) void flash_attn_wmma(
    const __bf16* __restrict__ qhat,  // [MQ, DIM], scaled by 1/tau
    const __bf16* __restrict__ khat,  // [MKV, DIM]
    const __bf16* __restrict__ vT,    // [B, H, HD, LKV]
    __bf16* __restrict__ ctx) {       // [MQ, DIM]
  __shared__ __bf16 pt[16][32];
  const int lane = threadIdx.x;
  const int r = lane & 15;
  const int kh = lane >> 4;
  const int m0 = blockIdx.x * 16;
  const int h = blockIdx.y;
  const int b = blockIdx.z;

  const __bf16* qbase =
      qhat + (size_t)(b * LQ + m0 + r) * DIMD + h * HD + kh * 8;
  const v16bf qA0 = ldfrag(qbase);        // K = 0..31  of head dim
  const v16bf qA1 = ldfrag(qbase + 32);   // K = 32..63

  v8f acc[4] = {};
  float mrun[8], lrun[8];
  #pragma unroll
  for (int e = 0; e < 8; ++e) { mrun[e] = -1e30f; lrun[e] = 0.f; }

  for (int kv0 = 0; kv0 < LKV; kv0 += 32) {
    // ---- scores: S[16 x 32] = q_tile(16x64) @ k_chunk(32x64)^T ----
    v8f s[2];
    #pragma unroll
    for (int u = 0; u < 2; ++u) {
      const __bf16* kb =
          khat + (size_t)(b * LKV + kv0 + u * 16 + r) * DIMD + h * HD + kh * 8;
      v16bf kB0 = ldfrag(kb);
      v16bf kB1 = ldfrag(kb + 32);
      v8f t = {};
      t = wmma_bf16(qA0, kB0, t);
      t = wmma_bf16(qA1, kB1, t);
      s[u] = t;
    }

    // ---- online softmax (row = e + 8*kh; row values across 16-lane group) --
    #pragma unroll
    for (int e = 0; e < 8; ++e) {
      float cm = fmaxf(s[0][e], s[1][e]);
      cm = fmaxf(cm, __shfl_xor(cm, 1, 32));
      cm = fmaxf(cm, __shfl_xor(cm, 2, 32));
      cm = fmaxf(cm, __shfl_xor(cm, 4, 32));
      cm = fmaxf(cm, __shfl_xor(cm, 8, 32));
      const float nm = fmaxf(mrun[e], cm);
      const float sc = expf(mrun[e] - nm);
      const float p0 = expf(s[0][e] - nm);
      const float p1 = expf(s[1][e] - nm);
      float rs = p0 + p1;
      rs += __shfl_xor(rs, 1, 32);
      rs += __shfl_xor(rs, 2, 32);
      rs += __shfl_xor(rs, 4, 32);
      rs += __shfl_xor(rs, 8, 32);
      lrun[e] = lrun[e] * sc + rs;
      mrun[e] = nm;
      #pragma unroll
      for (int u = 0; u < 4; ++u) acc[u][e] *= sc;
      s[0][e] = p0;
      s[1][e] = p1;
    }

    // ---- transpose P: C-fragment layout -> A-fragment layout via LDS ----
    __syncthreads();  // single-wave WG: S_NOP, keeps DS ordering explicit
    #pragma unroll
    for (int e = 0; e < 8; ++e) {
      pt[e + 8 * kh][r]      = (__bf16)s[0][e];
      pt[e + 8 * kh][16 + r] = (__bf16)s[1][e];
    }
    __syncthreads();
    const v16bf pA = ldfrag(&pt[r][kh * 8]);

    // ---- ctx[16 x 64] += P(16x32) @ V_chunk(32x64) ----
    #pragma unroll
    for (int u = 0; u < 4; ++u) {
      const __bf16* vb = vT +
          (size_t)((b * HEADS + h) * HD + u * 16 + r) * LKV + kv0 + kh * 8;
      v16bf vB = ldfrag(vb);
      acc[u] = wmma_bf16(pA, vB, acc[u]);
    }
    __syncthreads();
  }

  // ---- normalize and store ----
  #pragma unroll
  for (int e = 0; e < 8; ++e) {
    const float inv = 1.f / lrun[e];
    const int row = b * LQ + m0 + e + 8 * kh;
    #pragma unroll
    for (int u = 0; u < 4; ++u)
      ctx[(size_t)row * DIMD + h * HD + u * 16 + r] = (__bf16)(acc[u][e] * inv);
  }
}

// ---------------------------------------------------------------------------
// Host-side orchestration.
// ---------------------------------------------------------------------------
extern "C" void kernel_launch(void* const* d_in, const int* in_sizes, int n_in,
                              void* d_out, int out_size, void* d_ws,
                              size_t ws_size, hipStream_t stream) {
  (void)in_sizes; (void)n_in; (void)out_size; (void)ws_size;
  const float* q_tokens = (const float*)d_in[0];
  const float* kv_tokens = (const float*)d_in[1];
  const float* q_ln_w = (const float*)d_in[2];
  const float* q_ln_b = (const float*)d_in[3];
  const float* kv_ln_w = (const float*)d_in[4];
  const float* kv_ln_b = (const float*)d_in[5];
  const float* mlp_ln_w = (const float*)d_in[6];
  const float* mlp_ln_b = (const float*)d_in[7];
  const float* Wq = (const float*)d_in[8];
  const float* Wk = (const float*)d_in[9];
  const float* Wv = (const float*)d_in[10];
  const float* Wo = (const float*)d_in[11];
  const float* bo = (const float*)d_in[12];
  const float* fc1_w = (const float*)d_in[13];
  const float* fc1_b = (const float*)d_in[14];
  const float* fc2_w = (const float*)d_in[15];
  const float* fc2_b = (const float*)d_in[16];
  const float* alpha_attn = (const float*)d_in[17];
  const float* alpha_mlp = (const float*)d_in[18];
  float* out = (float*)d_out;

  // workspace carve (256B aligned)
  char* ws = (char*)d_ws;
  size_t off = 0;
  auto carve = [&](size_t bytes) -> char* {
    char* p = ws + off;
    off += (bytes + 255) & ~(size_t)255;
    return p;
  };
  __bf16* qn_bf   = (__bf16*)carve((size_t)MQ * DIMD * 2);
  __bf16* kvn_bf  = (__bf16*)carve((size_t)MKV * DIMD * 2);
  __bf16* Wq_bf   = (__bf16*)carve((size_t)DIMD * DIMD * 2);
  __bf16* Wk_bf   = (__bf16*)carve((size_t)DIMD * DIMD * 2);
  __bf16* Wv_bf   = (__bf16*)carve((size_t)DIMD * DIMD * 2);
  __bf16* Wo_bf   = (__bf16*)carve((size_t)DIMD * DIMD * 2);
  __bf16* fc1_bf  = (__bf16*)carve((size_t)HID * DIMD * 2);
  __bf16* fc2_bf  = (__bf16*)carve((size_t)DIMD * HID * 2);
  float*  q_f32   = (float*)carve((size_t)MQ * DIMD * 4);
  float*  k_f32   = (float*)carve((size_t)MKV * DIMD * 4);
  float*  v_f32   = (float*)carve((size_t)MKV * DIMD * 4);
  __bf16* qhat_bf = (__bf16*)carve((size_t)MQ * DIMD * 2);
  __bf16* khat_bf = (__bf16*)carve((size_t)MKV * DIMD * 2);
  __bf16* vT_bf   = (__bf16*)carve((size_t)MKV * DIMD * 2);
  __bf16* ctx_bf  = (__bf16*)carve((size_t)MQ * DIMD * 2);
  float*  att_out = (float*)carve((size_t)MQ * DIMD * 4);
  __bf16* hln_bf  = (__bf16*)carve((size_t)MQ * DIMD * 2);
  __bf16* g_bf    = (__bf16*)carve((size_t)MQ * HID * 2);

  // 1) LayerNorms -> bf16 activations
  ln_rows_bf16<<<MQ, 256, 0, stream>>>(q_tokens, q_ln_w, q_ln_b, qn_bf, DIMD);
  ln_rows_bf16<<<MKV, 256, 0, stream>>>(kv_tokens, kv_ln_w, kv_ln_b, kvn_bf, DIMD);

  // 2) Weight conversion fp32 -> bf16
  auto cvt = [&](const float* src, __bf16* dst, size_t n) {
    f32_to_bf16<<<(unsigned)((n + 255) / 256), 256, 0, stream>>>(src, dst, n);
  };
  cvt(Wq, Wq_bf, (size_t)DIMD * DIMD);
  cvt(Wk, Wk_bf, (size_t)DIMD * DIMD);
  cvt(Wv, Wv_bf, (size_t)DIMD * DIMD);
  cvt(Wo, Wo_bf, (size_t)DIMD * DIMD);
  cvt(fc1_w, fc1_bf, (size_t)HID * DIMD);
  cvt(fc2_w, fc2_bf, (size_t)DIMD * HID);

  // 3) Q/K/V projections (WMMA, double-buffered async-LDS staging)
  gemm_bf16_wmma<0><<<dim3(DIMD / 64, MQ / 64), 128, 0, stream>>>(
      qn_bf, Wq_bf, nullptr, nullptr, nullptr, q_f32, nullptr, MQ, DIMD, DIMD);
  gemm_bf16_wmma<0><<<dim3(DIMD / 64, MKV / 64), 128, 0, stream>>>(
      kvn_bf, Wk_bf, nullptr, nullptr, nullptr, k_f32, nullptr, MKV, DIMD, DIMD);
  gemm_bf16_wmma<0><<<dim3(DIMD / 64, MKV / 64), 128, 0, stream>>>(
      kvn_bf, Wv_bf, nullptr, nullptr, nullptr, v_f32, nullptr, MKV, DIMD, DIMD);

  // 4) L2-normalize head vectors (fold 1/tau into q), V transpose
  l2norm_bf16<<<(MQ * HEADS + 255) / 256, 256, 0, stream>>>(
      q_f32, qhat_bf, MQ * HEADS, 0.5f /* 1/TAU */);
  l2norm_bf16<<<(MKV * HEADS + 255) / 256, 256, 0, stream>>>(
      k_f32, khat_bf, MKV * HEADS, 1.0f);
  v_transpose_bf16<<<(BATCH * HEADS * HD * LKV + 255) / 256, 256, 0, stream>>>(
      v_f32, vT_bf);

  // 5) Flash attention (WMMA scores + WMMA P@V, online softmax)
  flash_attn_wmma<<<dim3(LQ / 16, HEADS, BATCH), 32, 0, stream>>>(
      qhat_bf, khat_bf, vT_bf, ctx_bf);

  // 6) Output projection + residual: att_out = q_tokens + a_attn*(ctx@Wo^T+bo)
  gemm_bf16_wmma<1><<<dim3(DIMD / 64, MQ / 64), 128, 0, stream>>>(
      ctx_bf, Wo_bf, bo, q_tokens, alpha_attn, att_out, nullptr, MQ, DIMD, DIMD);

  // 7) MLP: LN -> fc1+GELU -> fc2 + residual
  ln_rows_bf16<<<MQ, 256, 0, stream>>>(att_out, mlp_ln_w, mlp_ln_b, hln_bf, DIMD);
  gemm_bf16_wmma<2><<<dim3(HID / 64, MQ / 64), 128, 0, stream>>>(
      hln_bf, fc1_bf, fc1_b, nullptr, nullptr, nullptr, g_bf, MQ, HID, DIMD);
  gemm_bf16_wmma<1><<<dim3(DIMD / 64, MQ / 64), 128, 0, stream>>>(
      g_bf, fc2_bf, fc2_b, att_out, alpha_mlp, out, nullptr, MQ, DIMD, HID);
}
